// SwarmAttention_12979391168881
// MI455X (gfx1250) — compile-verified, matrix operand order checked by reading, not run on hardware
//
#include <hip/hip_runtime.h>
#include <math.h>

// ---------------- constants ----------------
#define NAG   256
#define BT    32
#define HID   512
#define HEADS 8
#define HD    64
#define MSGF  1024

typedef __attribute__((ext_vector_type(16))) _Float16 v16h;
typedef __attribute__((ext_vector_type(8)))  _Float16 h8;
typedef __attribute__((ext_vector_type(4)))  _Float16 h4;
typedef __attribute__((ext_vector_type(8)))  float    v8f;

// ---------------- WMMA helper ----------------
__device__ __forceinline__ v8f wmma32(v16h a, v16h b, v8f c) {
  // v_wmma_f32_16x16x32_f16: D = A(16x32) * B(32x16) + C(16x16)
  return __builtin_amdgcn_wmma_f32_16x16x32_f16(false, a, false, b, (short)0, c, false, false);
}

// A/B fragment from f16 memory. "row" = matrix row for A, matrix column for B
// (both K-contiguous in memory, stride ld elements).
// Lane l16 holds row/col, lane-half `hi` selects K sub-halves per ISA layout:
//   elem e in [0..7]  -> K = kb + hi*8 + e
//   elem e in [8..15] -> K = kb + 16 + hi*8 + (e-8)
__device__ __forceinline__ v16h frag_h16(const _Float16* base, int row, int ld,
                                         int kb, int hi) {
  const _Float16* p = base + (size_t)row * ld + kb + hi * 8;
  h8 a = *(const h8*)p;
  h8 b = *(const h8*)(p + 16);
  v16h f;
#pragma unroll
  for (int i = 0; i < 8; ++i) { f[i] = a[i]; f[i + 8] = b[i]; }
  return f;
}

// ---------------- K0a: zero the attn-weights output ----------------
__global__ void zero_f32(float* p, int n4) {
  int i = blockIdx.x * blockDim.x + threadIdx.x;
  if (i < n4) {
    float4 z = {0.f, 0.f, 0.f, 0.f};
    *(float4*)(p + (size_t)i * 4) = z;
  }
}

// ---------------- K0b: one-time f32 -> f16 conversion ----------------
__global__ void cvt_f16(const float* __restrict__ src, _Float16* __restrict__ dst, int n4) {
  int i = blockIdx.x * blockDim.x + threadIdx.x;
  if (i < n4) {
    float4 v = *(const float4*)(src + (size_t)i * 4);
    h4 o;
    o[0] = (_Float16)v.x; o[1] = (_Float16)v.y;
    o[2] = (_Float16)v.z; o[3] = (_Float16)v.w;
    *(h4*)(dst + (size_t)i * 4) = o;
  }
}

// ---------------- K1: species-gathered GEMM + LayerNorm ----------------
// One block per agent. A (f16) is (NAG, 32, K) contiguous per agent.
// Wh is (S, 512, K) f16, bias (S,512) f32. Out: f16 outH and/or f32 outF,
// layout (NAG, 32, 512).
__global__ __launch_bounds__(256) void species_ln(
    const _Float16* __restrict__ A, int K,
    const _Float16* __restrict__ Wh, const float* __restrict__ Bias,
    const float* __restrict__ gamma, const float* __restrict__ beta,
    const int* __restrict__ sidx, _Float16* outH, float* outF) {
  __shared__ float cbuf[BT * HID];  // 64 KB staging for LayerNorm

  const int n    = blockIdx.x;
  const int s    = sidx[n];
  const int tid  = threadIdx.x;
  const int wv   = tid >> 5;
  const int lane = tid & 31;
  const int l16  = lane & 15;
  const int hi   = lane >> 4;
  const int nb   = wv * 64;  // this wave's 64 output columns

  const _Float16* Wn = Wh + (size_t)s * HID * K;
  const float*    Bn = Bias + (size_t)s * HID;
  const _Float16* An = A + (size_t)n * BT * K;

  v8f acc[2][4] = {};
  for (int kb = 0; kb < K; kb += 32) {
    v16h a0 = frag_h16(An, l16, K, kb, hi);
    v16h a1 = frag_h16(An, 16 + l16, K, kb, hi);
#pragma unroll
    for (int t = 0; t < 4; ++t) {
      v16h bf = frag_h16(Wn, nb + t * 16 + l16, K, kb, hi);
      acc[0][t] = wmma32(a0, bf, acc[0][t]);
      acc[1][t] = wmma32(a1, bf, acc[1][t]);
    }
  }

  // bias add + stage to LDS
#pragma unroll
  for (int t = 0; t < 4; ++t) {
    const int o  = nb + t * 16 + l16;
    const float bv = Bn[o];
#pragma unroll
    for (int i = 0; i < 2; ++i) {
#pragma unroll
      for (int r = 0; r < 8; ++r) {
        const int m = i * 16 + r + hi * 8;
        cbuf[m * HID + o] = acc[i][t][r] + bv;
      }
    }
  }
  __syncthreads();

  // LayerNorm: 8 threads per row, 64 elems each
  const int row  = tid >> 3;
  const int part = tid & 7;
  float s1 = 0.f, s2 = 0.f;
#pragma unroll 8
  for (int j = 0; j < 64; ++j) {
    float x = cbuf[row * HID + part * 64 + j];
    s1 += x; s2 += x * x;
  }
#pragma unroll
  for (int m = 1; m < 8; m <<= 1) {
    s1 += __shfl_xor(s1, m, 32);
    s2 += __shfl_xor(s2, m, 32);
  }
  const float mu  = s1 * (1.f / (float)HID);
  const float var = s2 * (1.f / (float)HID) - mu * mu;
  const float inv = rsqrtf(var + 1e-5f);
  const size_t obase = ((size_t)n * BT + row) * HID;
#pragma unroll 8
  for (int j = 0; j < 64; ++j) {
    const int o = part * 64 + j;
    float y = (cbuf[row * HID + o] - mu) * inv * gamma[o] + beta[o];
    if (outH) outH[obase + o] = (_Float16)y;
    if (outF) outF[obase + o] = y;
  }
}

// ---------------- K2: generic projection GEMM with scatter epilogue --------
// C(M x 512) = A(f16, M x K, ld=K) @ Wh(f16, 512 x K)^T + bias.
// mode 0: A rows are (n,b) pairs  -> out[((b*8+h)*256+n)*64+d]   (qp/kp)
// mode 1: same rows               -> out[((b*8+h)*64+d)*256+n]   (vp transposed)
// mode 2: A rows are (b,n) pairs  -> out[(n*32+b)*512+o]         (attn_out -> (N,B,H))
__global__ __launch_bounds__(256) void proj_gemm(
    const _Float16* __restrict__ A, const _Float16* __restrict__ Wh,
    const float* __restrict__ Bias, _Float16* __restrict__ out, int K, int mode) {
  const int mb   = blockIdx.x * 32;
  const int tid  = threadIdx.x;
  const int wv   = tid >> 5;
  const int lane = tid & 31;
  const int l16  = lane & 15;
  const int hi   = lane >> 4;
  const int nb   = wv * 64;

  v8f acc[2][4] = {};
  for (int kb = 0; kb < K; kb += 32) {
    v16h a0 = frag_h16(A, mb + l16, K, kb, hi);
    v16h a1 = frag_h16(A, mb + 16 + l16, K, kb, hi);
#pragma unroll
    for (int t = 0; t < 4; ++t) {
      v16h bf = frag_h16(Wh, nb + t * 16 + l16, K, kb, hi);
      acc[0][t] = wmma32(a0, bf, acc[0][t]);
      acc[1][t] = wmma32(a1, bf, acc[1][t]);
    }
  }

#pragma unroll
  for (int t = 0; t < 4; ++t) {
    const int o  = nb + t * 16 + l16;
    const float bv = Bias[o];
#pragma unroll
    for (int i = 0; i < 2; ++i) {
#pragma unroll
      for (int r = 0; r < 8; ++r) {
        const int rg = mb + i * 16 + r + hi * 8;  // global row
        const float v = acc[i][t][r] + bv;
        size_t idx;
        if (mode == 0) {
          const int n = rg >> 5, b = rg & 31, h = o >> 6, d = o & 63;
          idx = (((size_t)b * HEADS + h) * NAG + n) * HD + d;
        } else if (mode == 1) {
          const int n = rg >> 5, b = rg & 31, h = o >> 6, d = o & 63;
          idx = (((size_t)b * HEADS + h) * HD + d) * NAG + n;
        } else {
          const int b = rg >> 8, n = rg & 255;
          idx = ((size_t)n * BT + b) * HID + o;
        }
        out[idx] = (_Float16)v;
      }
    }
  }
}

// ---------------- K3: masked attention per (b,h) ----------------
// grid (2, HEADS, BT), block 256 (8 waves). Each wave owns 16 query rows.
__global__ __launch_bounds__(256) void attention_kernel(
    const _Float16* __restrict__ qp, const _Float16* __restrict__ kp,
    const _Float16* __restrict__ vpT, const float* __restrict__ mask,
    _Float16* __restrict__ ctxbuf, float* __restrict__ out2) {
  __shared__ _Float16 sattn[8 * 16 * NAG];  // 64 KB: per-wave attn strip

  const int tid  = threadIdx.x;
  const int wv   = tid >> 5;
  const int lane = tid & 31;
  const int l16  = lane & 15;
  const int hi   = lane >> 4;
  const int b    = blockIdx.z;
  const int h    = blockIdx.y;
  const int rowbase = blockIdx.x * 128 + wv * 16;

  const _Float16* qb = qp  + ((size_t)b * HEADS + h) * NAG * HD;
  const _Float16* kb = kp  + ((size_t)b * HEADS + h) * NAG * HD;
  const _Float16* vb = vpT + ((size_t)b * HEADS + h) * HD * NAG;

  // scores: 16 rows x 256 cols, K = 64
  v8f s[16] = {};
  for (int kk = 0; kk < HD; kk += 32) {
    v16h a = frag_h16(qb, rowbase + l16, HD, kk, hi);
#pragma unroll
    for (int t = 0; t < 16; ++t) {
      v16h bf = frag_h16(kb, t * 16 + l16, HD, kk, hi);
      s[t] = wmma32(a, bf, s[t]);
    }
  }

  // masked softmax along the 256-wide row; row of reg r = rowbase + r + hi*8
#pragma unroll
  for (int r = 0; r < 8; ++r) {
    const int nq = rowbase + r + hi * 8;
    float mx = -__builtin_inff();
#pragma unroll
    for (int t = 0; t < 16; ++t) {
      float v = s[t][r] * 0.125f;  // 1/sqrt(64)
      const float mk = mask[(size_t)nq * NAG + t * 16 + l16];
      v = (mk > 0.5f) ? v : -__builtin_inff();
      s[t][r] = v;
      mx = fmaxf(mx, v);
    }
#pragma unroll
    for (int m = 1; m < 16; m <<= 1) mx = fmaxf(mx, __shfl_xor(mx, m, 32));
    float sum = 0.f;
#pragma unroll
    for (int t = 0; t < 16; ++t) {
      const float e = __expf(s[t][r] - mx);
      s[t][r] = e;
      sum += e;
    }
#pragma unroll
    for (int m = 1; m < 16; m <<= 1) sum += __shfl_xor(sum, m, 32);
    const float invs = 1.f / sum;
#pragma unroll
    for (int t = 0; t < 16; ++t) {
      const float a = s[t][r] * invs;
      const int nk = t * 16 + l16;
      atomicAdd(&out2[((size_t)b * NAG + nq) * NAG + nk], a * 0.125f);  // mean over 8 heads
      sattn[(wv * 16 + r + hi * 8) * NAG + nk] = (_Float16)a;
    }
  }
  __syncthreads();

  // ctx = attn(16x256) @ vp(256x64); B cols are K-contiguous rows of vpT
  const _Float16* sA = &sattn[wv * 16 * NAG];
  v8f c[4] = {};
  for (int kk = 0; kk < NAG; kk += 32) {
    v16h a = frag_h16(sA, l16, NAG, kk, hi);
#pragma unroll
    for (int t = 0; t < 4; ++t) {
      v16h bf = frag_h16(vb, t * 16 + l16, NAG, kk, hi);
      c[t] = wmma32(a, bf, c[t]);
    }
  }
#pragma unroll
  for (int t = 0; t < 4; ++t) {
#pragma unroll
    for (int r = 0; r < 8; ++r) {
      const int n  = rowbase + r + hi * 8;
      const int dd = t * 16 + l16;
      ctxbuf[((size_t)b * NAG + n) * HID + h * HD + dd] = (_Float16)c[t][r];
    }
  }
}

// ---------------- launch ----------------
extern "C" void kernel_launch(void* const* d_in, const int* in_sizes, int n_in,
                              void* d_out, int out_size, void* d_ws, size_t ws_size,
                              hipStream_t stream) {
  (void)in_sizes; (void)n_in; (void)out_size; (void)ws_size;

  const float* key_msgs   = (const float*)d_in[0];
  const int*   sidx       = (const int*)d_in[1];
  const float* query_msgs = (const float*)d_in[2];
  const float* mask       = (const float*)d_in[3];
  const float* sq_w = (const float*)d_in[4];  const float* sq_b = (const float*)d_in[5];
  const float* sk_w = (const float*)d_in[6];  const float* sk_b = (const float*)d_in[7];
  const float* sv_w = (const float*)d_in[8];  const float* sv_b = (const float*)d_in[9];
  const float* lnq_g = (const float*)d_in[10]; const float* lnq_b = (const float*)d_in[11];
  const float* lnk_g = (const float*)d_in[12]; const float* lnk_b = (const float*)d_in[13];
  const float* lnv_g = (const float*)d_in[14]; const float* lnv_b = (const float*)d_in[15];
  const float* lno_g = (const float*)d_in[16]; const float* lno_b = (const float*)d_in[17];
  const float* in_proj_w  = (const float*)d_in[18];
  const float* in_proj_b  = (const float*)d_in[19];
  const float* attn_out_w = (const float*)d_in[20];
  const float* attn_out_b = (const float*)d_in[21];
  const float* so_w = (const float*)d_in[22]; const float* so_b = (const float*)d_in[23];

  float* out  = (float*)d_out;                       // (N, B, H) = 4,194,304 f32
  float* out2 = out + (size_t)NAG * BT * HID;        // (B, N, N) = 2,097,152 f32

  // ---- workspace carve-up (all f16 buffers) ----
  char* ws = (char*)d_ws;
  size_t off = 0;
  auto take = [&](size_t elems) {
    _Float16* p = (_Float16*)(ws + off);
    off += elems * sizeof(_Float16);
    return p;
  };
  const size_t NBH = (size_t)NAG * BT * HID;   // 4,194,304
  _Float16* qh   = take(NBH);                  // (N,B,H)
  _Float16* kh   = take(NBH);
  _Float16* vh   = take(NBH);
  _Float16* qpb  = take(NBH);                  // (B,h,N,d)
  _Float16* kpb  = take(NBH);                  // (B,h,N,d)
  _Float16* vpT  = take(NBH);                  // (B,h,d,N)
  _Float16* ctxb = take(NBH);                  // (B,N,H)
  _Float16* aot  = take(NBH);                  // (N,B,H)
  // pre-converted f16 operands
  const size_t SW_BIG = (size_t)16 * HID * MSGF;   // 8,388,608 (sq/sk/sv_w)
  const size_t SW_SML = (size_t)16 * HID * HID;    // 4,194,304 (so_w)
  const size_t MSGN   = (size_t)NAG * BT * MSGF;   // 8,388,608 (msgs)
  _Float16* sqh = take(SW_BIG);
  _Float16* skh = take(SW_BIG);
  _Float16* svh = take(SW_BIG);
  _Float16* soh = take(SW_SML);
  _Float16* iph = take((size_t)3 * HID * HID);     // in_proj_w (1536x512)
  _Float16* aoh = take((size_t)HID * HID);         // attn_out_w
  _Float16* qmh = take(MSGN);                      // query_msgs f16
  _Float16* kmh = take(MSGN);                      // key_msgs f16

  auto cvt = [&](const float* s, _Float16* d, size_t n) {
    int n4 = (int)(n / 4);
    cvt_f16<<<(n4 + 255) / 256, 256, 0, stream>>>(s, d, n4);
  };

  // 0) zero attn-weights output (accumulated via atomics)
  zero_f32<<<2048, 256, 0, stream>>>(out2, (NAG * NAG * BT) / 4);

  // 1) one-time f32 -> f16 conversions (weights + messages)
  cvt(sq_w, sqh, SW_BIG);
  cvt(sk_w, skh, SW_BIG);
  cvt(sv_w, svh, SW_BIG);
  cvt(so_w, soh, SW_SML);
  cvt(in_proj_w, iph, (size_t)3 * HID * HID);
  cvt(attn_out_w, aoh, (size_t)HID * HID);
  cvt(query_msgs, qmh, MSGN);
  cvt(key_msgs, kmh, MSGN);

  // 2) species linears + LN -> f16 q/k/v
  species_ln<<<NAG, 256, 0, stream>>>(qmh, MSGF, sqh, sq_b, lnq_g, lnq_b, sidx, qh, nullptr);
  species_ln<<<NAG, 256, 0, stream>>>(kmh, MSGF, skh, sk_b, lnk_g, lnk_b, sidx, kh, nullptr);
  species_ln<<<NAG, 256, 0, stream>>>(kmh, MSGF, svh, sv_b, lnv_g, lnv_b, sidx, vh, nullptr);

  // 3) in_proj (Wq/Wk/Wv are row-blocks of in_proj_w)
  proj_gemm<<<(NAG * BT) / 32, 256, 0, stream>>>(qh, iph,              in_proj_b,        qpb, HID, 0);
  proj_gemm<<<(NAG * BT) / 32, 256, 0, stream>>>(kh, iph + 512 * 512,  in_proj_b + 512,  kpb, HID, 0);
  proj_gemm<<<(NAG * BT) / 32, 256, 0, stream>>>(vh, iph + 1024 * 512, in_proj_b + 1024, vpT, HID, 1);

  // 4) attention per (b,h): scores -> masked softmax -> ctx; head-mean to out2
  attention_kernel<<<dim3(2, HEADS, BT), 256, 0, stream>>>(qpb, kpb, vpT, mask, ctxb, out2);

  // 5) output projection, scattered to (N,B,H)
  proj_gemm<<<(NAG * BT) / 32, 256, 0, stream>>>(ctxb, aoh, attn_out_b, aot, HID, 2);

  // 6) final species linear + LN -> f32 output
  species_ln<<<NAG, 256, 0, stream>>>(aot, HID, soh, so_b, lno_g, lno_b, sidx, nullptr, out);
}
